// CostVolume_14078902796450
// MI455X (gfx1250) — matrix operand order repaired and checked by reading, not compiled, not run on hardware
//
#include <hip/hip_runtime.h>
#include <stdint.h>

// Windowed correlation cost volume on gfx1250 via bf16 WMMA.
// B=4, C=256, H=W=128, d=4 -> out [4, 81, 128, 128] fp32.

typedef __attribute__((ext_vector_type(16))) __bf16 v16bf;
typedef __attribute__((ext_vector_type(8)))  __bf16 v8bf;
typedef __attribute__((ext_vector_type(8)))  float  v8f;

#define Bn   4
#define Cn   256
#define Hn   128
#define Wn   128
#define HWn  (Hn * Wn)
#define NOFF 81

// fp32 -> bf16, round-to-nearest-even, returned in low 16 bits.
__device__ __forceinline__ unsigned int f2bf(float f) {
  unsigned int u = __float_as_uint(f);
  return ((u + 0x7FFFu + ((u >> 16) & 1u)) >> 16) & 0xFFFFu;
}

__global__ __launch_bounds__(256)
void cost_volume_wmma(const float* __restrict__ x1,
                      const float* __restrict__ x2,
                      float* __restrict__ out) {
  // Compute phase: A stage [64 w][32 c] bf16 = 4KB at offset 0,
  //                B stage [9 oi][80 col][32 c] bf16 = 45KB at offset 4096.
  // Epilogue (after barrier, reused): 36 D tiles of 16x16 f32 = 36KB.
  __shared__ __align__(16) unsigned char smem[50176];

  const int tid    = threadIdx.x;
  const int b      = blockIdx.z;
  const int h      = blockIdx.y;
  const int w_base = blockIdx.x * 64;

  const int lane = tid & 31;
  const int wid  = tid >> 5;    // wave id 0..7
  const int mrow = lane & 15;   // A row / B col within tile
  const int hl   = lane >> 4;   // lane half (k-group selector)

  v8f acc[9];
  const v8f vzero = {0.f, 0.f, 0.f, 0.f, 0.f, 0.f, 0.f, 0.f};
  #pragma unroll
  for (int i = 0; i < 9; ++i) acc[i] = vzero;

  for (int kc = 0; kc < 8; ++kc) {     // 8 chunks of 32 channels
    const int c0 = kc * 32;
    __syncthreads();                    // previous chunk's LDS reads done

    // ---- stage A: x1[b, c0..c0+31, h, w_base..w_base+63] -> LDS [w][c] bf16
    {
      const int w_idx = tid & 63;
      const int c_oct = tid >> 6;      // 0..3, 8 channels each
      const float* src = x1 + (((b * Cn + c0 + c_oct * 8) * Hn + h) * Wn)
                            + w_base + w_idx;
      if (kc + 1 < 8) __builtin_prefetch(src + 32 * HWn, 0, 0);
      unsigned int pk[4];
      #pragma unroll
      for (int e = 0; e < 4; ++e) {
        float f0 = src[(2 * e    ) * HWn];
        float f1 = src[(2 * e + 1) * HWn];
        pk[e] = f2bf(f0) | (f2bf(f1) << 16);
      }
      *(uint4*)(smem + w_idx * 64 + c_oct * 16) =
          make_uint4(pk[0], pk[1], pk[2], pk[3]);
    }

    // ---- stage B: x2 rows h-4..h+4, cols w_base-4..w_base+75 (zero-padded)
    //      -> LDS [oi*80+col][c] bf16 (fragment-contiguous in c)
    #pragma unroll
    for (int rep = 0; rep < 3; ++rep) {
      const int p = tid + rep * 256;   // pixel id: oi*80 + colIdx
      if (p < 720) {
        const int oi  = p / 80;
        const int ci  = p - oi * 80;
        const int row = h - 4 + oi;
        const int wc  = w_base - 4 + ci;
        const bool ok = ((unsigned)row < (unsigned)Hn) &&
                        ((unsigned)wc  < (unsigned)Wn);
        const float* src = x2 + (((b * Cn + c0) * Hn + (ok ? row : 0)) * Wn)
                              + (ok ? wc : 0);
        unsigned int pk[16];
        #pragma unroll
        for (int e = 0; e < 16; ++e) {
          float f0 = ok ? src[(2 * e    ) * HWn] : 0.f;
          float f1 = ok ? src[(2 * e + 1) * HWn] : 0.f;
          pk[e] = f2bf(f0) | (f2bf(f1) << 16);
        }
        uint4* dst = (uint4*)(smem + 4096 + p * 64);
        dst[0] = make_uint4(pk[0],  pk[1],  pk[2],  pk[3]);
        dst[1] = make_uint4(pk[4],  pk[5],  pk[6],  pk[7]);
        dst[2] = make_uint4(pk[8],  pk[9],  pk[10], pk[11]);
        dst[3] = make_uint4(pk[12], pk[13], pk[14], pk[15]);
      }
    }
    __syncthreads();

    // ---- compute: 9 WMMA jobs per wave; job j = (g, oi, col-tile t)
    const __bf16* Ab = (const __bf16*)smem;
    const __bf16* Bb = (const __bf16*)(smem + 4096);
    #pragma unroll
    for (int i = 0; i < 9; ++i) {
      const int j  = wid + 8 * i;      // 0..71
      const int g  = j / 18;           // w-group 0..3
      const int r  = j % 18;
      const int oi = r >> 1;           // row offset 0..8
      const int tt = g + (r & 1);      // col-tile 0..4
      // A fragment: lane = row (w_base+16g+mrow); k = hl*16 + e, contiguous
      const __bf16* ap = Ab + (16 * g + mrow) * 32 + hl * 16;
      v8bf lo = *(const v8bf*)(ap);
      v8bf hi = *(const v8bf*)(ap + 8);
      v16bf av = __builtin_shufflevector(lo, hi,
          0, 1, 2, 3, 4, 5, 6, 7, 8, 9, 10, 11, 12, 13, 14, 15);
      // B fragment: lane = col (tile tt, col mrow); k = hl*16 + e, contiguous
      v16bf bv = *(const v16bf*)(Bb + (oi * 80 + tt * 16 + mrow) * 32 + hl * 16);
      acc[i] = __builtin_amdgcn_wmma_f32_16x16x32_bf16(
          false, av, false, bv, (short)0, acc[i], false, false);
    }
  }

  // ---- epilogue: D tiles -> LDS, band-extract, coalesced stores.
  // Two rounds (g in {0,1} then {2,3}) keep LDS at 36KB.
  float* Dl = (float*)smem;
  #pragma unroll
  for (int hg = 0; hg < 2; ++hg) {
    __syncthreads();
    #pragma unroll
    for (int i = 0; i < 9; ++i) {
      const int j = wid + 8 * i;
      if (j >= 36 * hg && j < 36 * hg + 36) {
        const int jl = j - 36 * hg;
        #pragma unroll
        for (int r8 = 0; r8 < 8; ++r8)   // D: vgpr r -> m = r + 8*hl, n = mrow
          Dl[jl * 256 + (r8 + hl * 8) * 16 + mrow] = acc[i][r8];
      }
    }
    __syncthreads();
    for (int idx = tid; idx < 2592; idx += 256) {   // 81 offsets x 32 w
      const int o  = idx >> 5;
      const int wl = 32 * hg + (idx & 31);
      const int oi = o / 9;
      const int oj = o - oi * 9;
      const int g  = wl >> 4;
      const int m  = wl & 15;
      const int vc = wl + oj;                       // 16g + m + oj
      const int jl = (g - 2 * hg) * 18 + oi * 2 + ((vc >> 4) - g);
      out[((b * NOFF + o) * Hn + h) * Wn + w_base + wl] =
          Dl[jl * 256 + m * 16 + (vc & 15)] * (1.0f / 256.0f);
    }
  }
}

extern "C" void kernel_launch(void* const* d_in, const int* in_sizes, int n_in,
                              void* d_out, int out_size, void* d_ws, size_t ws_size,
                              hipStream_t stream) {
  const float* x1 = (const float*)d_in[0];
  const float* x2 = (const float*)d_in[1];
  float* out = (float*)d_out;
  (void)in_sizes; (void)n_in; (void)d_ws; (void)ws_size; (void)out_size;
  dim3 grid(Wn / 64, Hn, Bn);   // (2, 128, 4) = 1024 workgroups
  cost_volume_wmma<<<grid, 256, 0, stream>>>(x1, x2, out);
}
